// OutMultiChoices_transition_19292993093893
// MI455X (gfx1250) — matrix operand order repaired and checked by reading, not compiled, not run on hardware
//
#include <hip/hip_runtime.h>
#include <math.h>

#define TQ 40
#define TV 64
#define TA 40
#define BATCH 2048
#define DIM 1280
#define FEAT (4 * DIM)   // 5120
#define EMBED 512
#define BN_EPS 1e-5f

typedef __attribute__((ext_vector_type(16))) _Float16 v16h;
typedef __attribute__((ext_vector_type(8)))  float    v8f;
typedef __attribute__((ext_vector_type(4)))  float    vf4;
typedef __attribute__((ext_vector_type(4)))  _Float16 v4h;
typedef __attribute__((ext_vector_type(4)))  unsigned int vu4;

union H16 { vu4 u[2]; v16h h; };

// ---------------------------------------------------------------------------
// Phase 1: streaming masked mean over time, fp32 -> f16, writes a segment of
// feat[b, :]. Bandwidth-bound: b128 non-temporal loads, coalesced in D.
// Each thread owns 4 consecutive d's for one b.
// ---------------------------------------------------------------------------
__global__ __launch_bounds__(256)
void masked_mean_f16(const float* __restrict__ src, const int* __restrict__ lens,
                     int T, float invT, _Float16* __restrict__ feat, int seg_off) {
    int idx = blockIdx.x * 256 + threadIdx.x;     // covers BATCH*DIM/4 exactly
    int d4  = idx * 4;
    int b   = d4 / DIM;
    int d   = d4 - b * DIM;
    const size_t plane = (size_t)BATCH * DIM;
    const float* p = src + (size_t)b * DIM + d;

    int   n   = T;
    float inv = invT;
    if (lens) { n = lens[b]; inv = 1.0f / (float)n; }

    vf4 acc = {0.f, 0.f, 0.f, 0.f};
    for (int t = 0; t < n; ++t) {
        vf4 x = __builtin_nontemporal_load((const vf4*)(p + (size_t)t * plane));
        acc += x;
    }
    acc *= inv;

    v4h o;
    o[0] = (_Float16)acc[0]; o[1] = (_Float16)acc[1];
    o[2] = (_Float16)acc[2]; o[3] = (_Float16)acc[3];
    *(v4h*)(feat + (size_t)b * FEAT + seg_off + d) = o;
}

// ---------------------------------------------------------------------------
// W1 [FEAT, EMBED] fp32 -> W1t [EMBED, FEAT] f16 (LDS tile transpose so both
// the global read and the global write are coalesced).
// ---------------------------------------------------------------------------
__global__ __launch_bounds__(256)
void w1_transpose_f16(const float* __restrict__ W1, _Float16* __restrict__ W1t) {
    __shared__ float tile[32][33];
    int n0 = blockIdx.x * 32;          // EMBED direction
    int k0 = blockIdx.y * 32;          // FEAT direction
    int tx = threadIdx.x & 31;
    int ty = threadIdx.x >> 5;         // 0..7
    for (int i = 0; i < 4; ++i) {
        int k = k0 + ty + i * 8;
        tile[ty + i * 8][tx] = W1[(size_t)k * EMBED + n0 + tx];
    }
    __syncthreads();
    for (int i = 0; i < 4; ++i) {
        int n = n0 + ty + i * 8;
        W1t[(size_t)n * FEAT + k0 + tx] = (_Float16)tile[tx][ty + i * 8];
    }
}

// ---------------------------------------------------------------------------
// Phase 2: H = elu(feat @ W1 + b1) via v_wmma_f32_16x16x32_f16.
// Block = 8 waves (wave32), block tile 128(M) x 64(N), wave tile 32x32.
// Fragment loads are direct-from-global, contiguous per lane:
//   A (16x32 f16, row-major rows): lane(lo,hi) -> row m_base+lo,
//       halves [k0+8*hi .. +7] and [k0+16+8*hi .. +7]          (2x b128)
//   B (32x16 f16): from W1t[N,K] row-major -> lane(lo,hi) holds column
//       n_base+lo, halves [k0+8*hi .. +7] and [k0+16+8*hi .. +7] (2x b128)
// C/D layout: lane(lo,hi), vgpr r -> (M = 8*hi + r, N = lo).
// ---------------------------------------------------------------------------
__global__ __launch_bounds__(256, 1)
void gemm_wmma_elu(const _Float16* __restrict__ A,      // [BATCH, FEAT]
                   const _Float16* __restrict__ Bt,     // [EMBED, FEAT]
                   const float*    __restrict__ bias1,  // [EMBED]
                   float*          __restrict__ H) {    // [BATCH, EMBED]
    const int wave = threadIdx.x >> 5;
    const int lane = threadIdx.x & 31;
    const int lo   = lane & 15;
    const int hi   = lane >> 4;
    const int wm   = wave & 3;
    const int wn   = wave >> 2;
    const int m_base = blockIdx.x * 128 + wm * 32;
    const int n_base = blockIdx.y * 64  + wn * 32;

    v8f c00 = {0,0,0,0,0,0,0,0}, c01 = c00, c10 = c00, c11 = c00;

    const _Float16* arow0 = A  + (size_t)(m_base + lo)      * FEAT;
    const _Float16* arow1 = A  + (size_t)(m_base + 16 + lo) * FEAT;
    const _Float16* brow0 = Bt + (size_t)(n_base + lo)      * FEAT;
    const _Float16* brow1 = Bt + (size_t)(n_base + 16 + lo) * FEAT;

    for (int k0 = 0; k0 < FEAT; k0 += 32) {
        H16 fa0, fa1, fb0, fb1;
        fa0.u[0] = *(const vu4*)(arow0 + k0 + 8 * hi);
        fa0.u[1] = *(const vu4*)(arow0 + k0 + 16 + 8 * hi);
        fa1.u[0] = *(const vu4*)(arow1 + k0 + 8 * hi);
        fa1.u[1] = *(const vu4*)(arow1 + k0 + 16 + 8 * hi);
        fb0.u[0] = *(const vu4*)(brow0 + k0 + 8 * hi);
        fb0.u[1] = *(const vu4*)(brow0 + k0 + 16 + 8 * hi);
        fb1.u[0] = *(const vu4*)(brow1 + k0 + 8 * hi);
        fb1.u[1] = *(const vu4*)(brow1 + k0 + 16 + 8 * hi);

        c00 = __builtin_amdgcn_wmma_f32_16x16x32_f16(false, fa0.h, false, fb0.h,
                                                     (short)0, c00, false, false);
        c01 = __builtin_amdgcn_wmma_f32_16x16x32_f16(false, fa0.h, false, fb1.h,
                                                     (short)0, c01, false, false);
        c10 = __builtin_amdgcn_wmma_f32_16x16x32_f16(false, fa1.h, false, fb0.h,
                                                     (short)0, c10, false, false);
        c11 = __builtin_amdgcn_wmma_f32_16x16x32_f16(false, fa1.h, false, fb1.h,
                                                     (short)0, c11, false, false);
    }

    // Epilogue: + b1, ELU(alpha=1), store.
    v8f acc[2][2] = {{c00, c01}, {c10, c11}};
    for (int mi = 0; mi < 2; ++mi) {
        for (int ni = 0; ni < 2; ++ni) {
            int   n = n_base + ni * 16 + lo;
            float bb = bias1[n];
            for (int r = 0; r < 8; ++r) {
                int   m = m_base + mi * 16 + 8 * hi + r;
                float v = acc[mi][ni][r] + bb;
                v = (v > 0.f) ? v : (__expf(v) - 1.f);
                H[(size_t)m * EMBED + n] = v;
            }
        }
    }
}

// ---------------------------------------------------------------------------
// Phase 3a: per-column batch mean / rstd (biased var) over B=2048 rows.
// One block per 64 columns; 4 row-groups of 64 threads each, LDS reduce.
// ---------------------------------------------------------------------------
__global__ __launch_bounds__(256)
void bn_stats(const float* __restrict__ H, float* __restrict__ mu,
              float* __restrict__ rstd) {
    __shared__ float ssum[256], ssq[256];
    int c  = (threadIdx.x & 63) + blockIdx.x * 64;
    int rg = threadIdx.x >> 6;
    float s = 0.f, q = 0.f;
    for (int b = rg; b < BATCH; b += 4) {
        float x = H[(size_t)b * EMBED + c];
        s += x; q += x * x;
    }
    ssum[threadIdx.x] = s;
    ssq[threadIdx.x]  = q;
    __syncthreads();
    if (rg == 0) {
        for (int g = 1; g < 4; ++g) {
            s += ssum[threadIdx.x + 64 * g];
            q += ssq[threadIdx.x + 64 * g];
        }
        float m = s * (1.0f / BATCH);
        float v = q * (1.0f / BATCH) - m * m;
        mu[c]   = m;
        rstd[c] = rsqrtf(v + BN_EPS);
    }
}

// ---------------------------------------------------------------------------
// Phase 3b/4: out[b] = sum_e ((h[b,e]-mu)*rstd*gamma + beta) * W2[e] + b2.
// One wave32 per row, shuffle reduction.
// ---------------------------------------------------------------------------
__global__ __launch_bounds__(256)
void bn_gemv(const float* __restrict__ H, const float* __restrict__ mu,
             const float* __restrict__ rstd, const float* __restrict__ gamma,
             const float* __restrict__ beta, const float* __restrict__ W2,
             const float* __restrict__ b2, float* __restrict__ out) {
    int row  = (blockIdx.x * 256 + threadIdx.x) >> 5;
    int lane = threadIdx.x & 31;
    const float* h = H + (size_t)row * EMBED;
    float acc = 0.f;
    for (int i = 0; i < EMBED / 32; ++i) {
        int e = i * 32 + lane;
        float x = (h[e] - mu[e]) * rstd[e] * gamma[e] + beta[e];
        acc += x * W2[e];
    }
    for (int off = 16; off > 0; off >>= 1)
        acc += __shfl_down(acc, off, 32);
    if (lane == 0) out[row] = acc + b2[0];
}

// ---------------------------------------------------------------------------
extern "C" void kernel_launch(void* const* d_in, const int* in_sizes, int n_in,
                              void* d_out, int out_size, void* d_ws, size_t ws_size,
                              hipStream_t stream) {
    const float* q     = (const float*)d_in[0];
    const float* vq    = (const float*)d_in[1];
    const float* a     = (const float*)d_in[2];
    const float* va    = (const float*)d_in[3];
    const float* W1    = (const float*)d_in[4];
    const float* b1    = (const float*)d_in[5];
    const float* gamma = (const float*)d_in[6];
    const float* beta  = (const float*)d_in[7];
    const float* W2    = (const float*)d_in[8];
    const float* b2    = (const float*)d_in[9];
    const int*   qlen  = (const int*)d_in[10];
    const int*   alen  = (const int*)d_in[11];
    float* out = (float*)d_out;

    char* ws = (char*)d_ws;
    _Float16* feat = (_Float16*)(ws);                              // 20.0 MiB
    _Float16* w1t  = (_Float16*)(ws + (size_t)21 * 1024 * 1024);   //  5.0 MiB
    float*    H    = (float*)   (ws + (size_t)27 * 1024 * 1024);   //  4.0 MiB
    float*    mu   = (float*)   (ws + (size_t)32 * 1024 * 1024);
    float*    rstd = mu + EMBED;

    dim3 th(256);

    // Phase 1: four streaming means (2.18 GB, HBM-bound).
    int meanBlocks = (BATCH * DIM / 4) / 256;    // 2560
    masked_mean_f16<<<meanBlocks, th, 0, stream>>>(q,  qlen,    TQ, 0.f,        feat, 0);
    masked_mean_f16<<<meanBlocks, th, 0, stream>>>(vq, nullptr, TV, 1.f / TV,   feat, DIM);
    masked_mean_f16<<<meanBlocks, th, 0, stream>>>(a,  alen,    TA, 0.f,        feat, 2 * DIM);
    masked_mean_f16<<<meanBlocks, th, 0, stream>>>(va, nullptr, TV, 1.f / TV,   feat, 3 * DIM);

    // W1 -> f16 transposed [EMBED, FEAT].
    dim3 gT(EMBED / 32, FEAT / 32);              // 16 x 160
    w1_transpose_f16<<<gT, th, 0, stream>>>(W1, w1t);

    // Phase 2: WMMA GEMM + bias + ELU.
    dim3 gG(BATCH / 128, EMBED / 64);            // 16 x 8
    gemm_wmma_elu<<<gG, th, 0, stream>>>(feat, w1t, b1, H);

    // Phase 3: batch-norm stats, then normalize + GEMV + b2.
    bn_stats<<<EMBED / 64, th, 0, stream>>>(H, mu, rstd);
    bn_gemv<<<BATCH / 8, th, 0, stream>>>(H, mu, rstd, gamma, beta, W2, b2, out);
}
